// AttentionLayer_24464133718781
// MI455X (gfx1250) — matrix operand order repaired and checked by reading, not compiled, not run on hardware
//
#include <hip/hip_runtime.h>
#include <math.h>

#define Hdim 1024
#define Sdim 2048
#define Bdim 32
#define Udim 1024

typedef __attribute__((ext_vector_type(16))) __bf16 v16bf;
typedef __attribute__((ext_vector_type(8)))  float  v8f;

union V16 {
    v16bf    v;
    unsigned u[8];
    uint4    q[2];
};

// Hardware tanh if the toolchain exposes it (gfx1250 has V_TANH_F32),
// else a branch-free algebraic form: tanh(x) = 1 - 2/(e^{2x}+1).
// (Correct limits: e->inf => 1, e->0 => -1; no exec-mask divergence.)
__device__ __forceinline__ float fast_tanh(float x) {
#if __has_builtin(__builtin_amdgcn_tanhf)
    return __builtin_amdgcn_tanhf(x);
#elif __has_builtin(__builtin_amdgcn_tanh_f32)
    return __builtin_amdgcn_tanh_f32(x);
#else
    float e = __expf(2.0f * x);
    return 1.0f - 2.0f / (e + 1.0f);
#endif
}

// Accurate RNE pack (used once for W).
__device__ __forceinline__ unsigned pack_bf16x2_rne(float a, float b) {
    unsigned ua = __float_as_uint(a);
    unsigned ub = __float_as_uint(b);
    ua = (ua + 0x7fffu + ((ua >> 16) & 1u)) >> 16;
    ub = (ub + 0x7fffu + ((ub >> 16) & 1u)) >> 16;
    return (ua & 0xffffu) | (ub << 16);
}

// Fast pack for the streaming A operand: round-half-up bias then one
// v_perm_b32 grabs the two high halves. 3 VALU ops per bf16 pair,
// co-executes with XDL WMMA.
__device__ __forceinline__ unsigned pack_bf16x2_fast(float a, float b) {
    unsigned ua = __float_as_uint(a) + 0x8000u;
    unsigned ub = __float_as_uint(b) + 0x8000u;
    // dst bytes (lo..hi) = ua.b2, ua.b3, ub.b2, ub.b3
    return __builtin_amdgcn_perm(ub, ua, 0x07060302u);
}

// Load one lane's A-fragment (16-bit A 16x32 ISA layout):
// lane half h: elements 0..7  = K[h*8 .. h*8+7]
//              elements 8..15 = K[16+h*8 .. 16+h*8+7]
// p already points at (row, k + half*8) in f32.
__device__ __forceinline__ v16bf load_a_bf16(const float* __restrict__ p) {
    float4 f0 = *(const float4*)(p + 0);
    float4 f1 = *(const float4*)(p + 4);
    float4 f2 = *(const float4*)(p + 16);
    float4 f3 = *(const float4*)(p + 20);
    V16 r;
    r.u[0] = pack_bf16x2_fast(f0.x, f0.y);
    r.u[1] = pack_bf16x2_fast(f0.z, f0.w);
    r.u[2] = pack_bf16x2_fast(f1.x, f1.y);
    r.u[3] = pack_bf16x2_fast(f1.z, f1.w);
    r.u[4] = pack_bf16x2_fast(f2.x, f2.y);
    r.u[5] = pack_bf16x2_fast(f2.z, f2.w);
    r.u[6] = pack_bf16x2_fast(f3.x, f3.y);
    r.u[7] = pack_bf16x2_fast(f3.z, f3.w);
    return r.v;
}

// ---------------------------------------------------------------------------
// Kernel 1: pack W_enc = W_a[H:2H, :] (f32 row-major) into WMMA B-matrix
// bf16 layout. Tile (kt, nt) covers K=[kt*32, +32), N=[nt*16, +16).
// Per tile: 32 lanes x 8 dwords (16 bf16) contiguous -> 1 KiB.
// Lane L: N = nt*16 + (L&15); K = kt*32 + (L>>4)*8 + j (j<8), kt*32 + 16 + (L>>4)*8 + (j-8) (j>=8).
// ---------------------------------------------------------------------------
__global__ __launch_bounds__(256) void pack_w_kernel(const float* __restrict__ Wa,
                                                     unsigned* __restrict__ Wp) {
    int tid  = blockIdx.x * 256 + threadIdx.x;     // 2048 tiles * 32 lanes
    int lane = tid & 31;
    int tile = tid >> 5;                           // tile = kt*64 + nt
    int kt = tile >> 6;
    int nt = tile & 63;
    int kbase = kt * 32 + (lane >> 4) * 8;
    int n     = nt * 16 + (lane & 15);
    const float* src = Wa + (size_t)(Hdim + kbase) * Udim + n;
    unsigned o[8];
#pragma unroll
    for (int j = 0; j < 8; ++j)
        o[j] = pack_bf16x2_rne(src[(size_t)(2 * j + (j & 4) * 2) * Udim], src[(size_t)(2 * j + 1 + (j & 4) * 2) * Udim]);
    uint4* dst = (uint4*)(Wp + (size_t)tile * 256 + lane * 8);
    dst[0] = make_uint4(o[0], o[1], o[2], o[3]);
    dst[1] = make_uint4(o[4], o[5], o[6], o[7]);
}

// ---------------------------------------------------------------------------
// Kernel 2: dvec[b,u] = dec_h_t[b,:] @ W_a[:H, u] + b_a[u]
// ---------------------------------------------------------------------------
__global__ __launch_bounds__(256) void dec_proj_kernel(const float* __restrict__ dec,
                                                       const float* __restrict__ Wa,
                                                       const float* __restrict__ ba,
                                                       float* __restrict__ dvec) {
    int t = threadIdx.x;
    int b = t & 31;
    int u = blockIdx.x * 8 + (t >> 5);
    const float* dp = dec + (size_t)b * Hdim;
    const float* wp = Wa + u;
    float acc = 0.f;
#pragma unroll 8
    for (int h = 0; h < Hdim; ++h)
        acc += dp[h] * wp[(size_t)h * Udim];
    dvec[b * Udim + u] = acc + ba[u];
}

// ---------------------------------------------------------------------------
// Kernel 3 (hot): fused enc_proj GEMM + tanh + dot(v_a) -> scores[b,s]
// WG = 512 threads (16 waves), covers 32 rows (M) x N=1024.
// Wave w: BOTH M-tiles x 4 N-tiles [w*64, +64) -> each B tile loaded once
// and used by 2 WMMAs (halves B-side L0/L2 traffic per WMMA).
// K loop: 32 steps of 32; 8 v_wmma_f32_16x16x32_bf16 per wave per step.
// ---------------------------------------------------------------------------
__global__ __launch_bounds__(512) void attn_score_kernel(const float* __restrict__ enc,
                                                         const unsigned* __restrict__ Wp,
                                                         const float* __restrict__ dvec,
                                                         const float* __restrict__ va,
                                                         float* __restrict__ scores) {
    const int wg    = blockIdx.x;          // 0..2047
    const int rows0 = wg * 32;             // global row = b*S + s
    const int b     = rows0 >> 11;         // / Sdim
    const int s0    = rows0 & (Sdim - 1);
    const int wave  = threadIdx.x >> 5;    // 0..15 = N-group
    const int lane  = threadIdx.x & 31;
    const int half  = lane >> 4;
    const int l16   = lane & 15;
    const int ntb   = wave * 4;            // first N-tile index of this wave

    v8f c[2][4];
    const v8f vzero = {0.f, 0.f, 0.f, 0.f, 0.f, 0.f, 0.f, 0.f};
#pragma unroll
    for (int mt = 0; mt < 2; ++mt)
#pragma unroll
        for (int nt = 0; nt < 4; ++nt) c[mt][nt] = vzero;

    const float* arow0 = enc + ((size_t)b * Sdim + s0 + l16) * Hdim + half * 8;
    const float* arow1 = arow0 + (size_t)16 * Hdim;
    const unsigned* wlane = Wp + lane * 8;

    for (int k = 0; k < Hdim; k += 32) {
        v16bf a0 = load_a_bf16(arow0 + k);
        v16bf a1 = load_a_bf16(arow1 + k);
        const unsigned* tbase = wlane + (size_t)((k >> 5) * 64 + ntb) * 256;
#pragma unroll
        for (int nt = 0; nt < 4; ++nt) {
            const uint4* tp = (const uint4*)(tbase + (size_t)nt * 256);
            V16 bm;
            bm.q[0] = tp[0];
            bm.q[1] = tp[1];
            c[0][nt] = __builtin_amdgcn_wmma_f32_16x16x32_bf16(
                false, a0, false, bm.v, (short)0, c[0][nt], false, false);
            c[1][nt] = __builtin_amdgcn_wmma_f32_16x16x32_bf16(
                false, a1, false, bm.v, (short)0, c[1][nt], false, false);
        }
    }

    // tanh(enc_proj + dec_proj + b_a) * v_a, reduced over N.
    // Lane holds column n = (ntb+nt)*16 + l16 of rows (mt*16 + half*8 + j).
    float rp[2][8];
#pragma unroll
    for (int mt = 0; mt < 2; ++mt)
#pragma unroll
        for (int j = 0; j < 8; ++j) rp[mt][j] = 0.f;

#pragma unroll
    for (int nt = 0; nt < 4; ++nt) {
        int n    = (ntb + nt) * 16 + l16;
        float dv = dvec[b * Udim + n];
        float vv = va[n];
#pragma unroll
        for (int mt = 0; mt < 2; ++mt)
#pragma unroll
            for (int j = 0; j < 8; ++j)
                rp[mt][j] += fast_tanh(c[mt][nt][j] + dv) * vv;
    }

    __shared__ float red[16][32];
#pragma unroll
    for (int mt = 0; mt < 2; ++mt)
#pragma unroll
        for (int j = 0; j < 8; ++j) {
            float v = rp[mt][j];
            v += __shfl_xor(v, 1, 32);
            v += __shfl_xor(v, 2, 32);
            v += __shfl_xor(v, 4, 32);
            v += __shfl_xor(v, 8, 32);   // sum within each 16-lane half
            if (l16 == 0) red[wave][mt * 16 + half * 8 + j] = v;
        }
    __syncthreads();
    if (wave == 0) {
        // Every wave contributed a 64-column partial for every row; fixed order.
        int r = lane;
        float s = 0.f;
#pragma unroll
        for (int w = 0; w < 16; ++w) s += red[w][r];
        scores[(size_t)b * Sdim + s0 + r] = s;
    }
}

// ---------------------------------------------------------------------------
// Kernel 4: softmax over S per batch, in place. 256 threads x 8 values.
// ---------------------------------------------------------------------------
__global__ __launch_bounds__(256) void softmax_kernel(float* __restrict__ sc) {
    int b = blockIdx.x;
    float* p = sc + (size_t)b * Sdim;
    int t = threadIdx.x;
    __shared__ float lds[256];
    float v[8];
    float mx = -INFINITY;
#pragma unroll
    for (int i = 0; i < 8; ++i) {
        v[i] = p[t + i * 256];
        mx = fmaxf(mx, v[i]);
    }
    lds[t] = mx;
    __syncthreads();
    for (int s = 128; s > 0; s >>= 1) {
        if (t < s) lds[t] = fmaxf(lds[t], lds[t + s]);
        __syncthreads();
    }
    mx = lds[0];
    __syncthreads();
    float sum = 0.f;
#pragma unroll
    for (int i = 0; i < 8; ++i) {
        v[i] = __expf(v[i] - mx);
        sum += v[i];
    }
    lds[t] = sum;
    __syncthreads();
    for (int s = 128; s > 0; s >>= 1) {
        if (t < s) lds[t] += lds[t + s];
        __syncthreads();
    }
    float inv = 1.f / lds[0];
#pragma unroll
    for (int i = 0; i < 8; ++i) p[t + i * 256] = v[i] * inv;
}

// ---------------------------------------------------------------------------
// Kernel 5: partial context. Block (chunk, b): sum over 256 s of attn*enc.
// Thread t handles h = 4t..4t+3 (float4). part layout [B][8][H].
// ---------------------------------------------------------------------------
__global__ __launch_bounds__(256) void context_partial_kernel(const float* __restrict__ enc,
                                                              const float* __restrict__ attn,
                                                              float* __restrict__ part) {
    int chunk = blockIdx.x;   // 0..7
    int b     = blockIdx.y;   // 0..31
    int h     = threadIdx.x * 4;
    float4 acc = make_float4(0.f, 0.f, 0.f, 0.f);
    int sbeg = chunk * 256;
    const float* ep = enc + ((size_t)b * Sdim + sbeg) * Hdim + h;
    const float* ap = attn + (size_t)b * Sdim + sbeg;
    for (int s = 0; s < 256; ++s) {
        float a = ap[s];
        float4 e = *(const float4*)(ep + (size_t)s * Hdim);
        acc.x += a * e.x;
        acc.y += a * e.y;
        acc.z += a * e.z;
        acc.w += a * e.w;
    }
    *(float4*)(part + ((size_t)(b * 8 + chunk)) * Hdim + h) = acc;
}

__global__ __launch_bounds__(256) void context_reduce_kernel(const float* __restrict__ part,
                                                             float* __restrict__ out) {
    int i = blockIdx.x * 256 + threadIdx.x;  // 0..32*1024
    int b = i >> 10;
    int h = i & 1023;
    float s = 0.f;
#pragma unroll
    for (int c = 0; c < 8; ++c) s += part[((size_t)(b * 8 + c)) * Hdim + h];
    out[i] = s;
}

// ---------------------------------------------------------------------------
extern "C" void kernel_launch(void* const* d_in, const int* in_sizes, int n_in,
                              void* d_out, int out_size, void* d_ws, size_t ws_size,
                              hipStream_t stream) {
    (void)in_sizes; (void)n_in; (void)out_size; (void)ws_size;
    const float* dec = (const float*)d_in[0];  // [B,H]
    const float* enc = (const float*)d_in[1];  // [B,S,H]
    const float* Wa  = (const float*)d_in[2];  // [2H,U]
    const float* ba  = (const float*)d_in[3];  // [U]
    const float* va  = (const float*)d_in[4];  // [U,1]
    // d_in[5] = b_v: constant shift before softmax -> cancels; ignored.
    float* out = (float*)d_out;                // [B,H]

    char* ws = (char*)d_ws;
    unsigned* Wp = (unsigned*)ws;                                  // 2 MiB packed bf16 W
    float* dvec  = (float*)(ws + (2u << 20));                      // 128 KiB
    float* sc    = (float*)(ws + (2u << 20) + (128u << 10));       // 256 KiB scores/attn
    float* part  = (float*)(ws + (2u << 20) + (384u << 10));       // 1 MiB partials

    pack_w_kernel<<<256, 256, 0, stream>>>(Wa, Wp);
    dec_proj_kernel<<<Udim / 8, 256, 0, stream>>>(dec, Wa, ba, dvec);
    attn_score_kernel<<<(Bdim * Sdim) / 32, 512, 0, stream>>>(enc, Wp, dvec, va, sc);
    softmax_kernel<<<Bdim, 256, 0, stream>>>(sc);
    context_partial_kernel<<<dim3(8, Bdim), 256, 0, stream>>>(enc, sc, part);
    context_reduce_kernel<<<(Bdim * Hdim) / 256, 256, 0, stream>>>(part, out);
}
